// GPTJBlock_71055938945067
// MI455X (gfx1250) — matrix operand order repaired and checked by reading, not compiled, not run on hardware
//
#include <hip/hip_runtime.h>

// GPT-J block for MI455X (gfx1250, wave32, WMMA).
// Compute-bound (~1.8 TFLOP vs ~0.4GB bf16 weight traffic) -> bf16 WMMA everywhere,
// async global->LDS staging (ASYNCcnt) with double buffering where available.

#define B_ 2
#define S_ 2048
#define E_ 4096
#define H_ 16
#define D_ 256
#define ROT_ 64
#define BUCKET_ 2048
#define INNER_ 16384
#define MTOT_ (B_ * S_) // 4096 rows in all GEMMs

typedef unsigned short u16;
typedef __attribute__((ext_vector_type(16))) __bf16 v16bf;
typedef __attribute__((ext_vector_type(8)))  float  v8f;
typedef __attribute__((ext_vector_type(4)))  float  f4; // POD 16B vector for ld/st
typedef int v4i_ __attribute__((vector_size(16)));      // matches async-LDS builtin param

union Frag16 { f4 q[2]; v16bf v; }; // 32 bytes: one WMMA A/B operand per lane

__device__ __forceinline__ u16 f2bf(float x) {
  union { float f; unsigned u; } c; c.f = x;
  unsigned r = c.u + 0x7FFFu + ((c.u >> 16) & 1u);
  return (u16)(r >> 16);
}

__device__ __forceinline__ v8f wmma_bf16(v16bf a, v16bf b, v8f c) {
  // D = A(16x32) * B(32x16) + C, f32 accum
  return __builtin_amdgcn_wmma_f32_16x16x32_bf16(false, a, false, b, (short)0, c,
                                                 false, false);
}

// ---- async global->LDS (CDNA5 GLOBAL_LOAD_ASYNC_TO_LDS_B128, ASYNCcnt) ----
#if __has_builtin(__builtin_amdgcn_global_load_async_to_lds_b128)
#define ASYNC_LDS 1
#else
#define ASYNC_LDS 0
#endif

__device__ __forceinline__ void copy16_to_lds(const u16* g, u16* l) {
#if ASYNC_LDS
  __builtin_amdgcn_global_load_async_to_lds_b128(
      (__attribute__((address_space(1))) v4i_*)g,
      (__attribute__((address_space(3))) v4i_*)l, 0, 0);
#else
  *(f4*)l = *(const f4*)g;
#endif
}
__device__ __forceinline__ void async_wait_le4() {
#if ASYNC_LDS
#if __has_builtin(__builtin_amdgcn_s_wait_asynccnt)
  __builtin_amdgcn_s_wait_asynccnt(4);
#else
  asm volatile("s_wait_asynccnt 0x4" ::: "memory");
#endif
#endif
}
__device__ __forceinline__ void async_wait_0() {
#if ASYNC_LDS
#if __has_builtin(__builtin_amdgcn_s_wait_asynccnt)
  __builtin_amdgcn_s_wait_asynccnt(0);
#else
  asm volatile("s_wait_asynccnt 0x0" ::: "memory");
#endif
#endif
}

// ---------------------------------------------------------------- LayerNorm
__global__ __launch_bounds__(256) void ln_kernel(const float* __restrict__ x,
                                                 const float* __restrict__ g,
                                                 const float* __restrict__ be,
                                                 u16* __restrict__ hb) {
  int row = blockIdx.x;
  int t = threadIdx.x;
  const float* xr = x + (size_t)row * E_;
  float v[16];
  float s = 0.f, sq = 0.f;
#pragma unroll
  for (int j = 0; j < 16; ++j) {
    float xv = xr[t + 256 * j];
    v[j] = xv; s += xv; sq += xv * xv;
  }
#pragma unroll
  for (int m = 16; m >= 1; m >>= 1) {
    s  += __shfl_xor(s, m, 32);
    sq += __shfl_xor(sq, m, 32);
  }
  __shared__ float w1[8], w2[8];
  int wid = t >> 5, lane = t & 31;
  if (lane == 0) { w1[wid] = s; w2[wid] = sq; }
  __syncthreads();
  if (t == 0) {
    float a = 0.f, b2 = 0.f;
    for (int i = 0; i < 8; ++i) { a += w1[i]; b2 += w2[i]; }
    w1[0] = a; w2[0] = b2;
  }
  __syncthreads();
  float mu = w1[0] * (1.f / E_);
  float var = w2[0] * (1.f / E_) - mu * mu;
  float rs = rsqrtf(var + 1e-5f);
#pragma unroll
  for (int j = 0; j < 16; ++j) {
    int c = t + 256 * j;
    hb[(size_t)row * E_ + c] = f2bf((v[j] - mu) * rs * g[c] + be[c]);
  }
}

// --------------------------------------- Weight fp32 (K,N) -> bf16 (N,K) transposed
__global__ __launch_bounds__(256) void wconv_kernel(const float* __restrict__ W,
                                                    u16* __restrict__ Wt,
                                                    int K, int N) {
  __shared__ u16 tile[32][33];
  int k0 = blockIdx.y * 32, n0 = blockIdx.x * 32;
  int tx = threadIdx.x, ty = threadIdx.y;
#pragma unroll
  for (int i = 0; i < 4; ++i) {
    int k = ty + i * 8;
    tile[k][tx] = f2bf(W[(size_t)(k0 + k) * N + n0 + tx]);
  }
  __syncthreads();
#pragma unroll
  for (int i = 0; i < 4; ++i) {
    int n = ty + i * 8;
    Wt[(size_t)(n0 + n) * K + k0 + tx] = tile[tx][n];
  }
}

// ---------------------------------------------------------------- GEMM (bf16 WMMA)
// C(M,N) = A(M,K) @ W(K,N);  A row-major bf16, Wt = W^T (N,K) row-major bf16.
// Double-buffered LDS tiles with async global->LDS copies.
// epi: 0 -> store f32, 1 -> +bias, tanh-GELU, store bf16, 2 -> +bias, store f32.
__global__ __launch_bounds__(256) void gemm_kernel(const u16* __restrict__ A,
                                                   const u16* __restrict__ Wt,
                                                   const float* __restrict__ bias,
                                                   float* __restrict__ Cf,
                                                   u16* __restrict__ Cb,
                                                   int M, int N, int K, int epi) {
  __shared__ u16 As[2][128 * 32]; // 2 x 8 KB
  __shared__ u16 Bs[2][128 * 32]; // 2 x 8 KB (row = output col n, 32 contiguous k)
  int tid = threadIdx.x;
  int m0 = blockIdx.y * 128, n0 = blockIdx.x * 128;
  int w = tid >> 5, lane = tid & 31;
  int wm = w >> 2, wn = w & 3;       // 2x4 wave grid: 64 rows x 32 cols per wave
  int hlf = lane >> 4, l16 = lane & 15;
  int srow = tid >> 1, scol = (tid & 1) * 16; // staging: 16 bf16 per thread per tile

  const u16* ga0 = A + (size_t)(m0 + srow) * K + scol;
  const u16* gb0 = Wt + (size_t)(n0 + srow) * K + scol;

  auto stage = [&](int k0, int buf) {
    u16* la = &As[buf][srow * 32 + scol];
    u16* lb = &Bs[buf][srow * 32 + scol];
    copy16_to_lds(ga0 + k0, la);
    copy16_to_lds(ga0 + k0 + 8, la + 8);
    copy16_to_lds(gb0 + k0, lb);
    copy16_to_lds(gb0 + k0 + 8, lb + 8);
  };

  v8f acc[4][2] = {};
  int nk = K >> 5;
  stage(0, 0);
  for (int it = 0; it < nk; ++it) {
    int buf = it & 1;
    if (it + 1 < nk) {
      stage((it + 1) << 5, buf ^ 1); // prefetch next tile into other buffer
      async_wait_le4();              // current tile's 4 copies have retired
    } else {
      async_wait_0();
    }
    __syncthreads();
    Frag16 af[4];
#pragma unroll
    for (int mt = 0; mt < 4; ++mt) {
      const u16* p = &As[buf][(wm * 64 + mt * 16 + l16) * 32];
      af[mt].q[0] = *(const f4*)(p + hlf * 16);     // K 0..7 / 16..23
      af[mt].q[1] = *(const f4*)(p + hlf * 16 + 8); // K 8..15 / 24..31
    }
#pragma unroll
    for (int nt = 0; nt < 2; ++nt) {
      Frag16 bf;
      const u16* p = &Bs[buf][(wn * 32 + nt * 16 + l16) * 32 + hlf * 16];
      bf.q[0] = *(const f4*)p;
      bf.q[1] = *(const f4*)(p + 8);
#pragma unroll
      for (int mt = 0; mt < 4; ++mt)
        acc[mt][nt] = wmma_bf16(af[mt].v, bf.v, acc[mt][nt]);
    }
    __syncthreads(); // protect buf before it is refilled two tiles later
  }
#pragma unroll
  for (int mt = 0; mt < 4; ++mt) {
#pragma unroll
    for (int nt = 0; nt < 2; ++nt) {
      int col = n0 + wn * 32 + nt * 16 + l16;
      float bv = (epi != 0) ? bias[col] : 0.f;
#pragma unroll
      for (int r = 0; r < 8; ++r) {
        int row = m0 + wm * 64 + mt * 16 + r + 8 * hlf;
        float vv = acc[mt][nt][r] + bv;
        if (epi == 1) {
          float xx = vv;
          vv = 0.5f * xx *
               (1.f + tanhf(0.7978845608028654f * (xx + 0.044715f * xx * xx * xx)));
          Cb[(size_t)row * N + col] = f2bf(vv);
        } else {
          Cf[(size_t)row * N + col] = vv;
        }
      }
    }
  }
}

// ------------------------------------------------- KV-cache init from layer_past
__global__ __launch_bounds__(256) void cache_init_kernel(const float* __restrict__ past,
                                                         u16* __restrict__ cache) {
  size_t i = (size_t)blockIdx.x * 256 + threadIdx.x; // over B*H*BUCKET*D
  int d = (int)(i & 255);
  size_t r = i >> 8;
  int t = (int)(r & 2047); r >>= 11;
  int h = (int)(r & 15);
  int b = (int)(r >> 4);
  cache[i] = f2bf(past[(((size_t)b * BUCKET_ + t) * H_ + h) * D_ + d]);
}

// ---------------------------------------- RoPE + scatter into head-major bf16 bufs
__global__ __launch_bounds__(256) void rope_scatter_kernel(
    const float* __restrict__ qf, const float* __restrict__ kf,
    const float* __restrict__ vf, const int* __restrict__ pos,
    const int* __restrict__ kloc, const int* __restrict__ vloc,
    u16* __restrict__ qb, u16* __restrict__ kc, u16* __restrict__ vc) {
  int s = blockIdx.x, b = blockIdx.y, t = threadIdx.x;
  size_t rowoff = ((size_t)b * S_ + s) * E_;
  float p = (float)pos[b * S_ + s];
  int kl = kloc[b * S_ + s], vl = vloc[b * S_ + s];
#pragma unroll
  for (int j = 0; j < 8; ++j) {
    int e = 2 * (t + 256 * j); // pair base, 0..4094
    int h = e >> 8, d = e & 255;
    float q0 = qf[rowoff + e], q1 = qf[rowoff + e + 1];
    float k0 = kf[rowoff + e], k1 = kf[rowoff + e + 1];
    float v0 = vf[rowoff + e], v1 = vf[rowoff + e + 1];
    if (d < ROT_) {
      int i = d >> 1;
      float freq = __powf(10000.f, -((float)(2 * i)) / (float)ROT_);
      float ang = p * freq;
      float sn = __sinf(ang), cs = __cosf(ang);
      float a0 = q0 * cs - q1 * sn, a1 = q1 * cs + q0 * sn;
      float b0 = k0 * cs - k1 * sn, b1 = k1 * cs + k0 * sn;
      q0 = a0; q1 = a1; k0 = b0; k1 = b1;
    }
    size_t qo = (((size_t)b * H_ + h) * S_ + s) * D_ + d;
    qb[qo] = f2bf(q0); qb[qo + 1] = f2bf(q1);
    size_t ko = (((size_t)b * H_ + h) * BUCKET_ + kl) * D_ + d;
    kc[ko] = f2bf(k0); kc[ko + 1] = f2bf(k1);
    size_t vo = (((size_t)b * H_ + h) * BUCKET_ + vl) * D_ + d;
    vc[vo] = f2bf(v0); vc[vo + 1] = f2bf(v1);
  }
}

// ------------------------------------------------------- Flash attention (WMMA)
// Block: 128 threads (4 waves), one (b,h), 64 query rows; loops causal K blocks of 64.
// Q fragments register-resident; K staged async to LDS; V transposed into LDS; P via LDS.
__global__ __launch_bounds__(128) void flash_kernel(const u16* __restrict__ Qb,
                                                    const u16* __restrict__ Kc,
                                                    const u16* __restrict__ Vc,
                                                    u16* __restrict__ Ob) {
  extern __shared__ u16 smem[];
  u16* Ks = smem;                 // 64 x 256      (32 KB)
  u16* Vt = smem + 64 * 256;      // 256 x 64, d-major (32 KB)
  u16* Ps = smem + 2 * 64 * 256;  // 4 waves x 16 x 64 (8 KB)
  int tid = threadIdx.x;
  int w = tid >> 5, lane = tid & 31;
  int hlf = lane >> 4, l16 = lane & 15;
  int bh = blockIdx.y;
  int b = bh >> 4, h = bh & 15;
  int q0 = blockIdx.x * 64;
  const u16* Qbase = Qb + ((size_t)bh * S_) * D_;
  const u16* Kbase = Kc + ((size_t)bh * BUCKET_) * D_;
  const u16* Vbase = Vc + ((size_t)bh * BUCKET_) * D_;

  // Q A-fragments for this wave's 16 rows, all of D=256 (8 chunks of K=32).
  Frag16 qf[8];
  {
    const u16* qrow = Qbase + (size_t)(q0 + w * 16 + l16) * D_;
#pragma unroll
    for (int dd = 0; dd < 8; ++dd) {
      qf[dd].q[0] = *(const f4*)(qrow + dd * 32 + hlf * 16);
      qf[dd].q[1] = *(const f4*)(qrow + dd * 32 + hlf * 16 + 8);
    }
  }
  v8f O[16] = {};          // 16 rows x 256 cols f32 accum
  float mr[8], lr[8];
#pragma unroll
  for (int r = 0; r < 8; ++r) { mr[r] = -1e30f; lr[r] = 0.f; }

  int nkb = q0 / 64 + 1; // causal horizon in 64-key blocks
  for (int kb = 0; kb < nkb; ++kb) {
    // K tile: async memory->LDS copies (no VGPR round-trip)
#pragma unroll
    for (int j = 0; j < 16; ++j) {
      int cid = tid + 128 * j;   // 2048 8-element chunks
      int kr = cid >> 5;         // key row 0..63
      int kc8 = (cid & 31) * 8;  // d offset
      copy16_to_lds(Kbase + (size_t)(kb * 64 + kr) * D_ + kc8, Ks + kr * 256 + kc8);
    }
    // V tile: transpose through registers (d-major in LDS), overlaps the async Ks
#pragma unroll
    for (int j = 0; j < 16; ++j) {
      int cid = tid + 128 * j;
      int kr = cid >> 5;
      int kc8 = (cid & 31) * 8;
      f4 vv = *(const f4*)(Vbase + (size_t)(kb * 64 + kr) * D_ + kc8);
      const u16* vp = (const u16*)&vv;
#pragma unroll
      for (int i = 0; i < 8; ++i) Vt[(kc8 + i) * 64 + kr] = vp[i];
    }
    async_wait_0();
    __syncthreads();

    bool diag = (kb == nkb - 1);
    v8f st[4];
#pragma unroll
    for (int nt = 0; nt < 4; ++nt) {
      v8f s = {};
#pragma unroll
      for (int dd = 0; dd < 8; ++dd) {
        Frag16 bf;
        const u16* p = Ks + (nt * 16 + l16) * 256 + dd * 32 + hlf * 16;
        bf.q[0] = *(const f4*)p;
        bf.q[1] = *(const f4*)(p + 8);
        s = wmma_bf16(qf[dd].v, bf.v, s);
      }
      int kg = kb * 64 + nt * 16 + l16;
#pragma unroll
      for (int r = 0; r < 8; ++r) {
        float x = s[r] * 0.0625f; // 1/sqrt(256)
        if (diag) {
          int qr = q0 + w * 16 + r + 8 * hlf;
          if (kg > qr) x = -1e30f;
        }
        s[r] = x;
      }
      st[nt] = s;
    }
    // online softmax: rows live at (vgpr r, lane-half); reduce across 16-lane N group
    float mnew[8];
#pragma unroll
    for (int r = 0; r < 8; ++r) {
      float mx = fmaxf(fmaxf(st[0][r], st[1][r]), fmaxf(st[2][r], st[3][r]));
#pragma unroll
      for (int msk = 1; msk < 16; msk <<= 1) mx = fmaxf(mx, __shfl_xor(mx, msk, 32));
      mnew[r] = fmaxf(mr[r], mx);
      float alpha = __expf(mr[r] - mnew[r]);
      lr[r] *= alpha;
#pragma unroll
      for (int dt = 0; dt < 16; ++dt) O[dt][r] *= alpha;
      mr[r] = mnew[r];
    }
    u16* Pw = Ps + w * (16 * 64);
#pragma unroll
    for (int nt = 0; nt < 4; ++nt) {
#pragma unroll
      for (int r = 0; r < 8; ++r) {
        float pv = __expf(st[nt][r] - mnew[r]);
        float sum = pv;
#pragma unroll
        for (int msk = 1; msk < 16; msk <<= 1) sum += __shfl_xor(sum, msk, 32);
        lr[r] += sum; // this N-tile's contribution to the row sum
        Pw[(r + 8 * hlf) * 64 + nt * 16 + l16] = f2bf(pv);
      }
    }
    // P @ V
    Frag16 pf[2];
#pragma unroll
    for (int ks = 0; ks < 2; ++ks) {
      const u16* p = Pw + l16 * 64 + ks * 32;
      pf[ks].q[0] = *(const f4*)(p + hlf * 16);
      pf[ks].q[1] = *(const f4*)(p + hlf * 16 + 8);
    }
#pragma unroll
    for (int dt = 0; dt < 16; ++dt) {
#pragma unroll
      for (int ks = 0; ks < 2; ++ks) {
        Frag16 bf;
        const u16* p = Vt + (dt * 16 + l16) * 64 + ks * 32 + hlf * 16;
        bf.q[0] = *(const f4*)p;
        bf.q[1] = *(const f4*)(p + 8);
        O[dt] = wmma_bf16(pf[ks].v, bf.v, O[dt]);
      }
    }
    __syncthreads();
  }
#pragma unroll
  for (int r = 0; r < 8; ++r) lr[r] = 1.f / lr[r];
#pragma unroll
  for (int dt = 0; dt < 16; ++dt) {
    int d = dt * 16 + l16;
#pragma unroll
    for (int r = 0; r < 8; ++r) {
      int sr = q0 + w * 16 + r + 8 * hlf;
      Ob[((size_t)b * S_ + sr) * E_ + h * D_ + d] = f2bf(O[dt][r] * lr[r]);
    }
  }
}

// ---------------------------------------------------------------- final residual
__global__ __launch_bounds__(256) void add3_kernel(const float* __restrict__ a,
                                                   const float* __restrict__ bb,
                                                   const float* __restrict__ c,
                                                   float* __restrict__ o, size_t n) {
  size_t i = (size_t)blockIdx.x * 256 + threadIdx.x;
  if (i < n) o[i] = a[i] + bb[i] + c[i];
}

extern "C" void kernel_launch(void* const* d_in, const int* in_sizes, int n_in,
                              void* d_out, int out_size, void* d_ws, size_t ws_size,
                              hipStream_t stream) {
  (void)in_sizes; (void)n_in; (void)out_size; (void)ws_size;
  const float* hidden   = (const float*)d_in[0];
  const float* ln_g     = (const float*)d_in[1];
  const float* ln_b     = (const float*)d_in[2];
  const float* wq       = (const float*)d_in[3];
  const float* wk       = (const float*)d_in[4];
  const float* wv       = (const float*)d_in[5];
  const float* wo       = (const float*)d_in[6];
  const float* fc_in_w  = (const float*)d_in[7];
  const float* fc_in_b  = (const float*)d_in[8];
  const float* fc_out_w = (const float*)d_in[9];
  const float* fc_out_b = (const float*)d_in[10];
  const float* past_k   = (const float*)d_in[11];
  const float* past_v   = (const float*)d_in[12];
  const int*   kloc     = (const int*)d_in[13];
  const int*   vloc     = (const int*)d_in[14];
  const int*   pos      = (const int*)d_in[15];
  float* out = (float*)d_out;

  char* ws = (char*)d_ws;
  size_t off = 0;
  auto alloc = [&](size_t bytes) -> void* {
    void* p = ws + off;
    off += (bytes + 255) & ~(size_t)255;
    return p;
  };
  u16* h_bf   = (u16*)alloc((size_t)MTOT_ * E_ * 2);
  u16* wqT    = (u16*)alloc((size_t)E_ * E_ * 2);
  u16* wkT    = (u16*)alloc((size_t)E_ * E_ * 2);
  u16* wvT    = (u16*)alloc((size_t)E_ * E_ * 2);
  u16* woT    = (u16*)alloc((size_t)E_ * E_ * 2);
  u16* fciT   = (u16*)alloc((size_t)INNER_ * E_ * 2); // (N=INNER, K=E)
  u16* fcoT   = (u16*)alloc((size_t)E_ * INNER_ * 2); // (N=E, K=INNER)
  float* q_f  = (float*)alloc((size_t)MTOT_ * E_ * 4);
  float* k_f  = (float*)alloc((size_t)MTOT_ * E_ * 4);
  float* v_f  = (float*)alloc((size_t)MTOT_ * E_ * 4);
  u16* qb     = (u16*)alloc((size_t)B_ * H_ * S_ * D_ * 2);
  u16* kcB    = (u16*)alloc((size_t)B_ * H_ * BUCKET_ * D_ * 2);
  u16* vcB    = (u16*)alloc((size_t)B_ * H_ * BUCKET_ * D_ * 2);
  u16* ob     = (u16*)alloc((size_t)MTOT_ * E_ * 2);
  float* attn = (float*)alloc((size_t)MTOT_ * E_ * 4);
  u16* act    = (u16*)alloc((size_t)MTOT_ * INNER_ * 2);
  float* ff   = (float*)alloc((size_t)MTOT_ * E_ * 4);

  // 1) LayerNorm -> bf16
  ln_kernel<<<MTOT_, 256, 0, stream>>>(hidden, ln_g, ln_b, h_bf);

  // 2) weight convert+transpose
  dim3 tb(32, 8);
  wconv_kernel<<<dim3(E_ / 32, E_ / 32), tb, 0, stream>>>(wq, wqT, E_, E_);
  wconv_kernel<<<dim3(E_ / 32, E_ / 32), tb, 0, stream>>>(wk, wkT, E_, E_);
  wconv_kernel<<<dim3(E_ / 32, E_ / 32), tb, 0, stream>>>(wv, wvT, E_, E_);
  wconv_kernel<<<dim3(E_ / 32, E_ / 32), tb, 0, stream>>>(wo, woT, E_, E_);
  wconv_kernel<<<dim3(INNER_ / 32, E_ / 32), tb, 0, stream>>>(fc_in_w, fciT, E_, INNER_);
  wconv_kernel<<<dim3(E_ / 32, INNER_ / 32), tb, 0, stream>>>(fc_out_w, fcoT, INNER_, E_);

  // 3) QKV projections (f32 out)
  dim3 gEE(E_ / 128, MTOT_ / 128);
  gemm_kernel<<<gEE, 256, 0, stream>>>(h_bf, wqT, nullptr, q_f, nullptr, MTOT_, E_, E_, 0);
  gemm_kernel<<<gEE, 256, 0, stream>>>(h_bf, wkT, nullptr, k_f, nullptr, MTOT_, E_, E_, 0);
  gemm_kernel<<<gEE, 256, 0, stream>>>(h_bf, wvT, nullptr, v_f, nullptr, MTOT_, E_, E_, 0);

  // 4) cache init + RoPE scatter
  int cacheN = B_ * H_ * BUCKET_ * D_;
  cache_init_kernel<<<cacheN / 256, 256, 0, stream>>>(past_k, kcB);
  cache_init_kernel<<<cacheN / 256, 256, 0, stream>>>(past_v, vcB);
  rope_scatter_kernel<<<dim3(S_, B_), 256, 0, stream>>>(q_f, k_f, v_f, pos, kloc, vloc,
                                                        qb, kcB, vcB);

  // 5) flash attention
  size_t fa_lds = (size_t)(2 * 64 * 256 + 4 * 16 * 64) * sizeof(u16); // 72 KB
  flash_kernel<<<dim3(S_ / 64, B_ * H_), 128, fa_lds, stream>>>(qb, kcB, vcB, ob);

  // 6) output projection, FFN, residual
  gemm_kernel<<<gEE, 256, 0, stream>>>(ob, woT, nullptr, attn, nullptr, MTOT_, E_, E_, 0);
  gemm_kernel<<<dim3(INNER_ / 128, MTOT_ / 128), 256, 0, stream>>>(
      h_bf, fciT, fc_in_b, nullptr, act, MTOT_, INNER_, E_, 1);
  gemm_kernel<<<gEE, 256, 0, stream>>>(act, fcoT, fc_out_b, ff, nullptr, MTOT_, E_, INNER_, 2);

  size_t n = (size_t)MTOT_ * E_;
  add3_kernel<<<(unsigned)((n + 255) / 256), 256, 0, stream>>>(attn, ff, hidden, out, n);
}
